// Ours_att_74062416053444
// MI455X (gfx1250) — compile-verified, matrix operand order checked by reading, not run on hardware
//
#include <hip/hip_runtime.h>
#include <math.h>
#include <stdint.h>

// Problem sizes (match reference)
#define DV   16384   // V
#define DE   8192    // E
#define DD   128     // D
#define DK   16      // K (nodes per hyperedge)
#define DKN  8       // KN (edges per node)
#define DM   16      // M inducing points
#define DH   256     // H mlp hidden

typedef __attribute__((ext_vector_type(16))) __bf16 v16bf;
typedef __attribute__((ext_vector_type(8)))  float  v8f;
typedef __attribute__((ext_vector_type(4)))  unsigned int u32x4;
typedef __attribute__((ext_vector_type(8)))  int          i32x8;
typedef __attribute__((ext_vector_type(4)))  int          i32x4;

struct MabW { const float *Wq,*bq,*Wk,*bk,*Wv,*bv,*Wo,*bo; };

// ---------------------------------------------------------------------------
// WMMA helpers (CDNA5, wave32, 16x16x32 bf16 -> f32)
// ---------------------------------------------------------------------------
__device__ __forceinline__ v8f wmma_bf16(v16bf a, v16bf b, v8f c) {
  // 8 args: (neg_a, A, neg_b, B, c_mod, C, reuse_a, reuse_b)
  return __builtin_amdgcn_wmma_f32_16x16x32_bf16(false, a, false, b,
                                                 (short)0, c, false, false);
}

__device__ __forceinline__ v8f zero8() {
  v8f a;
  #pragma unroll
  for (int i = 0; i < 8; ++i) a[i] = 0.0f;
  return a;
}

__device__ __forceinline__ v8f splat8(float x) {
  v8f a;
  #pragma unroll
  for (int i = 0; i < 8; ++i) a[i] = x;
  return a;
}

// A-fragment (16xK tile, row-major f32 source, convert to bf16).
// Documented 16-bit A layout: lanes 0-15 rows M=0..15 (kgroup 0),
// lanes 16-31 same rows kgroup 1; VGPR v<4 -> K base v*2 (+kgroup*8),
// v>=4 -> K base 16+(v-4)*2 (+kgroup*8).
__device__ __forceinline__ v16bf load_a_frag(const float* s, int ld, int kb, int kmax) {
  const int lane = threadIdx.x & 31;
  const int r    = lane & 15;
  const int kg   = (lane >> 4) << 3;
  v16bf f;
  #pragma unroll
  for (int v = 0; v < 8; ++v) {
    const int base = ((v < 4) ? 0 : 16) + kg + ((v & 3) << 1);
    const int k0 = kb + base;
    float x0 = (k0     < kmax) ? s[r * ld + k0    ] : 0.0f;
    float x1 = (k0 + 1 < kmax) ? s[r * ld + k0 + 1] : 0.0f;
    f[2 * v]     = (__bf16)x0;
    f[2 * v + 1] = (__bf16)x1;
  }
  return f;
}

// B-fragment (Kx16 tile from row-major [K x N] f32 source), N<->M mirror of A.
__device__ __forceinline__ v16bf load_b_frag(const float* s, int ld, int kb, int cb, int kmax) {
  const int lane = threadIdx.x & 31;
  const int c    = cb + (lane & 15);
  const int kg   = (lane >> 4) << 3;
  v16bf f;
  #pragma unroll
  for (int v = 0; v < 8; ++v) {
    const int base = ((v < 4) ? 0 : 16) + kg + ((v & 3) << 1);
    const int k0 = kb + base;
    float x0 = (k0     < kmax) ? s[(size_t)(k0    ) * ld + c] : 0.0f;
    float x1 = (k0 + 1 < kmax) ? s[(size_t)(k0 + 1) * ld + c] : 0.0f;
    f[2 * v]     = (__bf16)x0;
    f[2 * v + 1] = (__bf16)x1;
  }
  return f;
}

// B-fragment where B[k][c] = src[c*ld + k]  (i.e. B = src^T, for Q @ K^T)
__device__ __forceinline__ v16bf load_bT_frag(const float* s, int ld, int kb, int cb) {
  const int lane = threadIdx.x & 31;
  const int c    = cb + (lane & 15);
  const int kg   = (lane >> 4) << 3;
  v16bf f;
  #pragma unroll
  for (int v = 0; v < 8; ++v) {
    const int base = ((v < 4) ? 0 : 16) + kg + ((v & 3) << 1);
    const int k0 = kb + base;
    f[2 * v]     = (__bf16)s[c * ld + k0];
    f[2 * v + 1] = (__bf16)s[c * ld + k0 + 1];
  }
  return f;
}

// C/D layout: lane n = lane&15, VGPR v -> row v + ((lane>>4)<<3)
__device__ __forceinline__ void store_acc(float* s, int ld, int cb, v8f acc) {
  const int lane = threadIdx.x & 31;
  const int n    = cb + (lane & 15);
  const int mo   = (lane >> 4) << 3;
  #pragma unroll
  for (int v = 0; v < 8; ++v) s[(mo + v) * ld + n] = acc[v];
}

__device__ __forceinline__ v8f load_res(const float* s, int ld, int cb) {
  const int lane = threadIdx.x & 31;
  const int n    = cb + (lane & 15);
  const int mo   = (lane >> 4) << 3;
  v8f a;
  #pragma unroll
  for (int v = 0; v < 8; ++v) a[v] = s[(mo + v) * ld + n];
  return a;
}

__device__ __forceinline__ float wave_sum(float x) {
  #pragma unroll
  for (int m = 16; m; m >>= 1) x += __shfl_xor(x, m, 32);
  return x;
}

#define KBIG (1 << 30)

// ---------------------------------------------------------------------------
// Tensor Data Mover: build a 2D-tile D# (CDNA5 ISA) and
// issue TENSOR_LOAD_TO_LDS. This toolchain exposes the 6-arg builtin form:
//   (uint32x4 g0, int32x8 g1, int32x4 g2, int32x4 g3, int32x8 gx, i32 cpol)
//   tile = tileRows x tileCols f32 elements, row stride strideElems,
//   DMA'd contiguously into LDS at byte offset ldsOff.
// ---------------------------------------------------------------------------
__device__ __forceinline__ unsigned lds_addr_of(const void* p) {
  return (unsigned)(unsigned long long)(uintptr_t)p;   // LDS offset = low 32 bits of the pointer
}

__device__ __forceinline__ void tdm_load_tile_f32(const float* gsrc, unsigned ldsOff,
                                                  int tileCols, int tileRows,
                                                  int strideElems) {
  unsigned long long ga = (unsigned long long)(uintptr_t)gsrc;
  u32x4 g0;
  g0[0] = 1u;                                          // count=1, user D#
  g0[1] = ldsOff;                                      // lds_addr (bytes)
  g0[2] = (unsigned)ga;                                // global_addr[31:0]
  g0[3] = (unsigned)((ga >> 32) & 0x01ffffffu)         // global_addr[56:32]
        | (2u << 30);                                  // type = 2 ("image")
  i32x8 g1;
  g1[0] = 2 << 16;                                     // data_size = 4 bytes
  g1[1] = (tileCols & 0xffff) << 16;                   // tensor_dim0[15:0]
  g1[2] = ((tileCols >> 16) & 0xffff)                  // tensor_dim0[31:16]
        | ((tileRows & 0xffff) << 16);                 // tensor_dim1[15:0]
  g1[3] = ((tileRows >> 16) & 0xffff)                  // tensor_dim1[31:16]
        | ((tileCols & 0xffff) << 16);                 // tile_dim0
  g1[4] = (tileRows & 0xffff);                         // tile_dim1 (tile_dim2=0)
  g1[5] = strideElems;                                 // tensor_dim0_stride[31:0]
  g1[6] = 0;                                           // stride hi / dim1 stride
  g1[7] = 0;
  i32x4 z4; z4[0] = z4[1] = z4[2] = z4[3] = 0;         // 2D: groups 2/3 unused
  i32x8 z8;
  #pragma unroll
  for (int i = 0; i < 8; ++i) z8[i] = 0;
  __builtin_amdgcn_tensor_load_to_lds(g0, g1, z4, z4, z8, 0);
}

// ---------------------------------------------------------------------------
// Per-wave MAB building blocks (one wave32 per batch element, tiles in LDS)
// ---------------------------------------------------------------------------

// sDst[16x128] = sXsrc[16x128] @ W[128x128] + bias
__device__ __forceinline__ void wave_linear(const float* sXsrc, const float* W,
                                            const float* bias, float* sDst) {
  const int lane = threadIdx.x & 31;
  const int nl   = lane & 15;
  #pragma unroll
  for (int nt = 0; nt < 8; ++nt) {
    const int cb = nt * 16;
    v8f acc = splat8(bias[cb + nl]);
    #pragma unroll
    for (int kb = 0; kb < DD; kb += 32) {
      v16bf a = load_a_frag(sXsrc, DD, kb, KBIG);
      v16bf b = load_b_frag(W, DD, kb, cb, KBIG);
      acc = wmma_bf16(a, b, acc);
    }
    store_acc(sDst, DD, cb, acc);
  }
}

// row-wise softmax of sS[16x16]: rows >= nv zeroed, cols >= mv zeroed.
__device__ __forceinline__ void wave_softmax(float* sS, int nv, int mv, float scale) {
  const int lane = threadIdx.x & 31;
  if (lane < 16) {
    float* row = sS + lane * 16;
    if (lane < nv) {
      float mx = -1e30f;
      for (int j = 0; j < mv; ++j) { float x = row[j] * scale; row[j] = x; mx = fmaxf(mx, x); }
      float sum = 0.0f;
      for (int j = 0; j < mv; ++j) { float e = __expf(row[j] - mx); row[j] = e; sum += e; }
      const float inv = 1.0f / sum;
      for (int j = 0; j < 16; ++j) row[j] = (j < mv) ? row[j] * inv : 0.0f;
    } else {
      for (int j = 0; j < 16; ++j) row[j] = 0.0f;
    }
  }
}

// Set-transformer MAB (queries pre-projected):
//   S = softmax(Q K^T / sqrt(D)), O = Q + S V, Out = O + relu(O Wo + bo)
__device__ __forceinline__ void mab_wave(const float* sQ, const float* sY,
                                         int nv, int mv, const MabW& w,
                                         float* sK, float* sV, float* sS,
                                         float* sO, float* sOut) {
  wave_linear(sY, w.Wk, w.bk, sK); __syncthreads();
  wave_linear(sY, w.Wv, w.bv, sV); __syncthreads();

  { // scores
    v8f acc = zero8();
    #pragma unroll
    for (int kb = 0; kb < DD; kb += 32) {
      v16bf a = load_a_frag(sQ, DD, kb, KBIG);
      v16bf b = load_bT_frag(sK, DD, kb, 0);
      acc = wmma_bf16(a, b, acc);
    }
    store_acc(sS, 16, 0, acc);
  }
  __syncthreads();
  wave_softmax(sS, nv, mv, 0.08838834764831845f);  // 1/sqrt(128)
  __syncthreads();

  // O = Q + S V   (single 16x16x32 WMMA per n-tile, K padded with zeros)
  #pragma unroll
  for (int nt = 0; nt < 8; ++nt) {
    v8f acc = load_res(sQ, DD, nt * 16);
    v16bf a = load_a_frag(sS, 16, 0, 16);
    v16bf b = load_b_frag(sV, DD, 0, nt * 16, 16);
    acc = wmma_bf16(a, b, acc);
    store_acc(sO, DD, nt * 16, acc);
  }
  __syncthreads();

  // Out = O + relu(O Wo + bo); zero padded query rows
  const int lane = threadIdx.x & 31;
  const int nl   = lane & 15;
  const int mo   = (lane >> 4) << 3;
  #pragma unroll
  for (int nt = 0; nt < 8; ++nt) {
    const int cb = nt * 16;
    const int n  = cb + nl;
    v8f acc = splat8(w.bo[n]);
    #pragma unroll
    for (int kb = 0; kb < DD; kb += 32) {
      v16bf a = load_a_frag(sO, DD, kb, KBIG);
      v16bf b = load_b_frag(w.Wo, DD, kb, cb, KBIG);
      acc = wmma_bf16(a, b, acc);
    }
    #pragma unroll
    for (int v = 0; v < 8; ++v) {
      const int m = mo + v;
      const float val = sO[m * DD + n] + fmaxf(acc[v], 0.0f);
      sOut[m * DD + n] = (m < nv) ? val : 0.0f;
    }
  }
  __syncthreads();
}

// ---------------------------------------------------------------------------
// Generic WMMA GEMM: C = act(A[MxK] @ B[KxN] + bias); fp32 global, bf16 MACs.
// Block = 256 threads (8 waves), tile 128x128, K-step 32 staged via LDS.
// ---------------------------------------------------------------------------
#define BM 128
#define BN 128
#define BK 32

__global__ __launch_bounds__(256) void k_gemm(
    const float* __restrict__ A, int lda,
    const float* __restrict__ B, int ldb,
    const float* __restrict__ bias,
    float* __restrict__ C, int ldc,
    int M, int N, int Kdim, int act) {
  __shared__ __align__(16) float sA[BM * BK];
  __shared__ __align__(16) float sB[BK * BN];
  const int tid  = threadIdx.x;
  const int wave = tid >> 5;
  const int lane = tid & 31;
  const int row0 = blockIdx.x * BM;
  const int col0 = blockIdx.y * BN;

  v8f acc[8];
  #pragma unroll
  for (int t = 0; t < 8; ++t) acc[t] = zero8();

  for (int kb = 0; kb < Kdim; kb += BK) {
    #pragma unroll
    for (int i = 0; i < 16; ++i) {           // A tile 128x32
      int idx = i * 256 + tid;
      int r = idx >> 5, c = idx & 31;
      int gr = row0 + r, gk = kb + c;
      sA[idx] = (gr < M && gk < Kdim) ? A[(size_t)gr * lda + gk] : 0.0f;
    }
    #pragma unroll
    for (int i = 0; i < 16; ++i) {           // B tile 32x128
      int idx = i * 256 + tid;
      int r = idx >> 7, c = idx & 127;
      int gk = kb + r, gc = col0 + c;
      sB[idx] = (gk < Kdim && gc < N) ? B[(size_t)gk * ldb + gc] : 0.0f;
    }
    if (kb + BK < Kdim) {                    // gfx1250 global_prefetch_b8
      int pr = row0 + (tid >> 1);
      if (pr < M) __builtin_prefetch(&A[(size_t)pr * lda + kb + BK], 0, 0);
      __builtin_prefetch(&B[(size_t)(kb + BK + (tid >> 4)) * ldb + col0 + (tid & 15) * 8], 0, 0);
    }
    __syncthreads();
    v16bf afr = load_a_frag(sA + wave * 16 * BK, BK, 0, BK);
    #pragma unroll
    for (int nt = 0; nt < 8; ++nt) {
      v16bf bfr = load_b_frag(sB, BN, 0, nt * 16, BK);
      acc[nt] = wmma_bf16(afr, bfr, acc[nt]);
    }
    __syncthreads();
  }

  const int nl = lane & 15;
  const int mo = (lane >> 4) << 3;
  #pragma unroll
  for (int nt = 0; nt < 8; ++nt) {
    int gc = col0 + nt * 16 + nl;
    float bb = (bias != nullptr && gc < N) ? bias[gc] : 0.0f;
    #pragma unroll
    for (int v = 0; v < 8; ++v) {
      int gr = row0 + wave * 16 + mo + v;
      if (gr < M && gc < N) {
        float x = acc[nt][v] + bb;
        if (act == 1) x = fmaxf(x, 0.0f);
        C[(size_t)gr * ldc + gc] = x;
      }
    }
  }
}

// ---------------------------------------------------------------------------
// TDM-fed big-K GEMM for the dense adjacency products:
//   C[M x 128] = A[M x K] @ B[K x 128]   (fp32 global, bf16 WMMA, f32 accum)
// Tiles DMA'd into LDS by the Tensor Data Mover with double buffering:
// wave 0 issues descriptors for tile t+1, waits TENSORcnt<=2 (tile t known
// complete, two newest in flight), barrier publishes LDS to all 4 waves.
// ---------------------------------------------------------------------------
#define TBM 64
#define TBK 32

__global__ __launch_bounds__(128) void k_gemm_tdm(
    const float* __restrict__ A, int lda,
    const float* __restrict__ B,                 // [K x 128], ldb = 128
    float* __restrict__ C,                       // [M x 128]
    int M, int Kdim) {
  __shared__ __align__(16) float sA[2][TBM * TBK];
  __shared__ __align__(16) float sB[2][TBK * DD];
  const int tid  = threadIdx.x;
  const int wave = tid >> 5;
  const int lane = tid & 31;
  const int row0 = blockIdx.x * TBM;

  v8f acc[8];
  #pragma unroll
  for (int t = 0; t < 8; ++t) acc[t] = zero8();

  const int nT = Kdim / TBK;
  if (tid < 32) {                                // one wave drives the TDM
    tdm_load_tile_f32(&A[(size_t)row0 * lda], lds_addr_of(&sA[0][0]),
                      TBK, TBM, lda);
    tdm_load_tile_f32(B, lds_addr_of(&sB[0][0]), DD, TBK, DD);
  }

  for (int t = 0; t < nT; ++t) {
    if (tid < 32) {
      if (t + 1 < nT) {
        tdm_load_tile_f32(&A[(size_t)row0 * lda + (size_t)(t + 1) * TBK],
                          lds_addr_of(&sA[(t + 1) & 1][0]), TBK, TBM, lda);
        tdm_load_tile_f32(&B[(size_t)(t + 1) * TBK * DD],
                          lds_addr_of(&sB[(t + 1) & 1][0]), DD, TBK, DD);
        __builtin_amdgcn_s_wait_tensorcnt(2);    // tile t complete
      } else {
        __builtin_amdgcn_s_wait_tensorcnt(0);
      }
    }
    __syncthreads();
    const float* cA = &sA[t & 1][0];
    const float* cB = &sB[t & 1][0];
    v16bf afr = load_a_frag(cA + wave * 16 * TBK, TBK, 0, TBK);
    #pragma unroll
    for (int nt = 0; nt < 8; ++nt) {
      v16bf bfr = load_b_frag(cB, DD, 0, nt * 16, TBK);
      acc[nt] = wmma_bf16(afr, bfr, acc[nt]);
    }
    __syncthreads();                             // buffer reuse fence
  }

  const int nl = lane & 15;
  const int mo = (lane >> 4) << 3;
  #pragma unroll
  for (int nt = 0; nt < 8; ++nt) {
    const int gc = nt * 16 + nl;
    #pragma unroll
    for (int v = 0; v < 8; ++v) {
      const int gr = row0 + wave * 16 + mo + v;
      C[(size_t)gr * DD + gc] = acc[nt][v];
    }
  }
}

// ---------------------------------------------------------------------------
// Fused edge encoder: one wave32 per hyperedge.
// ---------------------------------------------------------------------------
__global__ __launch_bounds__(32) void k_edge_encode(
    const float* __restrict__ featv, const float* __restrict__ efeat,
    const int* __restrict__ edge_set, const float* __restrict__ q0,
    MabW m0, MabW m1, MabW ve,
    const float* __restrict__ lng, const float* __restrict__ lnb,
    float* __restrict__ feate) {
  __shared__ __align__(16) float sX[16 * DD];
  __shared__ __align__(16) float sQ[16 * DD];
  __shared__ __align__(16) float sK[16 * DD];
  __shared__ __align__(16) float sV[16 * DD];
  __shared__ __align__(16) float sO[16 * DD];
  __shared__ __align__(16) float sH[16 * DD];
  __shared__ __align__(16) float sA[16 * DD];
  __shared__ __align__(16) float sS[16 * 16];

  const int e    = blockIdx.x;
  const int lane = threadIdx.x;

  #pragma unroll
  for (int t = 0; t < 16; ++t) {
    int idx = t * 32 + lane;
    int r = idx >> 5, c4 = idx & 31;
    int src = edge_set[e * DK + r];
    ((float4*)sX)[idx] = ((const float4*)(featv + (size_t)src * DD))[c4];
    ((float4*)sQ)[idx] = ((const float4*)q0)[idx];
  }
  __syncthreads();

  mab_wave(sQ, sX, 16, 16, m0, sK, sV, sS, sO, sH);        // H = MAB(I, Xn)

  wave_linear(sX, m1.Wq, m1.bq, sQ); __syncthreads();
  mab_wave(sQ, sH, 16, 16, m1, sK, sV, sS, sO, sA);        // att = MAB(Xn, H)

  #pragma unroll
  for (int t = 0; t < 16; ++t) {
    int idx = t * 32 + lane;
    int r = idx >> 5, c4 = idx & 31;
    float4 z; z.x = z.y = z.z = z.w = 0.0f;
    ((float4*)sX)[idx] = (r == 0) ? ((const float4*)(efeat + (size_t)e * DD))[c4] : z;
  }
  __syncthreads();
  wave_linear(sX, ve.Wq, ve.bq, sQ); __syncthreads();
  mab_wave(sQ, sA, 1, 16, ve, sK, sV, sS, sO, sH);         // fe in sH row 0

  float x[4]; float sum = 0.0f;
  #pragma unroll
  for (int i = 0; i < 4; ++i) { x[i] = sH[lane * 4 + i]; sum += x[i]; }
  sum = wave_sum(sum);
  const float mu = sum * (1.0f / 128.0f);
  float vs = 0.0f;
  #pragma unroll
  for (int i = 0; i < 4; ++i) { float d = x[i] - mu; vs += d * d; }
  vs = wave_sum(vs);
  const float inv = rsqrtf(vs * (1.0f / 128.0f) + 1e-5f);
  #pragma unroll
  for (int i = 0; i < 4; ++i) {
    int c = lane * 4 + i;
    feate[(size_t)e * DD + c] = (x[i] - mu) * inv * lng[c] + lnb[c];
  }
}

// ---------------------------------------------------------------------------
// Fused node decoder: one wave32 per vertex (KN=8 incident edges, zero-padded)
// ---------------------------------------------------------------------------
__global__ __launch_bounds__(32) void k_node_decode(
    const float* __restrict__ featv, const float* __restrict__ fe2,
    const int* __restrict__ node_set, const float* __restrict__ q0,
    MabW m0, MabW m1, MabW ev,
    const float* __restrict__ lng, const float* __restrict__ lnb,
    float* __restrict__ out) {
  __shared__ __align__(16) float sX[16 * DD];
  __shared__ __align__(16) float sQ[16 * DD];
  __shared__ __align__(16) float sK[16 * DD];
  __shared__ __align__(16) float sV[16 * DD];
  __shared__ __align__(16) float sO[16 * DD];
  __shared__ __align__(16) float sH[16 * DD];
  __shared__ __align__(16) float sA[16 * DD];
  __shared__ __align__(16) float sS[16 * 16];

  const int vtx  = blockIdx.x;
  const int lane = threadIdx.x;

  #pragma unroll
  for (int t = 0; t < 16; ++t) {
    int idx = t * 32 + lane;
    int r = idx >> 5, c4 = idx & 31;
    float4 val; val.x = val.y = val.z = val.w = 0.0f;
    if (r < DKN) {
      int src = node_set[vtx * DKN + r];
      val = ((const float4*)(fe2 + (size_t)src * DD))[c4];
    }
    ((float4*)sX)[idx] = val;
    ((float4*)sQ)[idx] = ((const float4*)q0)[idx];
  }
  __syncthreads();

  mab_wave(sQ, sX, 16, DKN, m0, sK, sV, sS, sO, sH);       // H = MAB(I, Xe)

  wave_linear(sX, m1.Wq, m1.bq, sQ); __syncthreads();
  mab_wave(sQ, sH, DKN, 16, m1, sK, sV, sS, sO, sA);       // att_edge (8 rows)

  #pragma unroll
  for (int t = 0; t < 16; ++t) {
    int idx = t * 32 + lane;
    int r = idx >> 5, c4 = idx & 31;
    float4 z; z.x = z.y = z.z = z.w = 0.0f;
    ((float4*)sX)[idx] = (r == 0) ? ((const float4*)(featv + (size_t)vtx * DD))[c4] : z;
  }
  __syncthreads();
  wave_linear(sX, ev.Wq, ev.bq, sQ); __syncthreads();
  mab_wave(sQ, sA, 1, DKN, ev, sK, sV, sS, sO, sH);        // row 0 result

  float x[4]; float sum = 0.0f;
  #pragma unroll
  for (int i = 0; i < 4; ++i) { x[i] = sH[lane * 4 + i]; sum += x[i]; }
  sum = wave_sum(sum);
  const float mu = sum * (1.0f / 128.0f);
  float vs = 0.0f;
  #pragma unroll
  for (int i = 0; i < 4; ++i) { float d = x[i] - mu; vs += d * d; }
  vs = wave_sum(vs);
  const float inv = rsqrtf(vs * (1.0f / 128.0f) + 1e-5f);
  #pragma unroll
  for (int i = 0; i < 4; ++i) {
    int c = lane * 4 + i;
    out[(size_t)vtx * DD + c] = (x[i] - mu) * inv * lng[c] + lnb[c];
  }
}

// ---------------------------------------------------------------------------
// Small glue kernels
// ---------------------------------------------------------------------------
__global__ __launch_bounds__(128) void k_layernorm(
    const float* __restrict__ X, const float* __restrict__ g,
    const float* __restrict__ b, float* __restrict__ Y) {
  __shared__ float red[128];
  const int row = blockIdx.x;
  const int c   = threadIdx.x;
  float x = X[(size_t)row * DD + c];
  red[c] = x; __syncthreads();
  for (int s = 64; s > 0; s >>= 1) { if (c < s) red[c] += red[c + s]; __syncthreads(); }
  float mu = red[0] * (1.0f / 128.0f); __syncthreads();
  float d = x - mu;
  red[c] = d * d; __syncthreads();
  for (int s = 64; s > 0; s >>= 1) { if (c < s) red[c] += red[c + s]; __syncthreads(); }
  float var = red[0] * (1.0f / 128.0f);
  Y[(size_t)row * DD + c] = d * rsqrtf(var + 1e-5f) * g[c] + b[c];
}

__global__ __launch_bounds__(256) void k_concat(
    const float* __restrict__ s, const float* __restrict__ t,
    float* __restrict__ cat) {
  int i = blockIdx.x * blockDim.x + threadIdx.x;
  if (i >= DE * DD) return;
  int e = i >> 7, c = i & 127;
  cat[(size_t)e * (2 * DD) + c]      = s[i];
  cat[(size_t)e * (2 * DD) + DD + c] = t[i];
}

__global__ __launch_bounds__(64) void k_zero(float* __restrict__ p, int n) {
  int i = blockIdx.x * blockDim.x + threadIdx.x;
  if (i < n) p[i] = 0.0f;
}

// One wave per edge: accumulate the three cosine sums.
__global__ __launch_bounds__(256) void k_cos(
    const float* __restrict__ s, const float* __restrict__ t,
    float* __restrict__ accum) {
  const int wave = blockIdx.x * (blockDim.x >> 5) + (threadIdx.x >> 5);
  const int lane = threadIdx.x & 31;
  if (wave >= DE) return;
  const int e  = wave;
  const int p1 = (e * 2531 + 7)   & (DE - 1);   // deterministic odd-mult perms
  const int p2 = (e * 4099 + 101) & (DE - 1);
  float dst = 0, ds1 = 0, dt2 = 0, ns = 0, nt = 0, ns1 = 0, nt2 = 0;
  for (int i = lane; i < DD; i += 32) {
    float a = s[(size_t)e  * DD + i], b = t[(size_t)e  * DD + i];
    float c = s[(size_t)p1 * DD + i], d = t[(size_t)p2 * DD + i];
    dst += a * b; ds1 += a * c; dt2 += b * d;
    ns  += a * a; nt  += b * b; ns1 += c * c; nt2 += d * d;
  }
  dst = wave_sum(dst); ds1 = wave_sum(ds1); dt2 = wave_sum(dt2);
  ns = wave_sum(ns); nt = wave_sum(nt); ns1 = wave_sum(ns1); nt2 = wave_sum(nt2);
  if (lane == 0) {
    const float eps = 1e-8f;
    atomicAdd(&accum[0], dst / fmaxf(sqrtf(ns) * sqrtf(nt),  eps));
    atomicAdd(&accum[1], ds1 / fmaxf(sqrtf(ns) * sqrtf(ns1), eps));
    atomicAdd(&accum[2], dt2 / fmaxf(sqrtf(nt) * sqrtf(nt2), eps));
  }
}

__global__ __launch_bounds__(32) void k_loss(const float* __restrict__ accum,
                                             float* __restrict__ out) {
  if (threadIdx.x != 0) return;
  const float invE = 1.0f / (float)DE;
  float pos = accum[0] * invE, n1 = accum[1] * invE, n2 = accum[2] * invE;
  float sp  = 1.0f / (1.0f + expf(-pos));
  float s1  = 1.0f / (1.0f + expf(-n1));
  float s2  = 1.0f / (1.0f + expf(-n2));
  out[DV * DD] = logf(sp) + logf(1.0f - s1) + logf(1.0f - s2);
}

// ---------------------------------------------------------------------------
// Host launch
// ---------------------------------------------------------------------------
extern "C" void kernel_launch(void* const* d_in, const int* in_sizes, int n_in,
                              void* d_out, int out_size, void* d_ws, size_t ws_size,
                              hipStream_t stream) {
  (void)in_sizes; (void)n_in; (void)out_size; (void)ws_size;
  int p = 0;
  auto nf = [&]() { return (const float*)d_in[p++]; };
  auto nmab = [&]() {
    MabW w;
    w.Wq = nf(); w.bq = nf(); w.Wk = nf(); w.bk = nf();
    w.Wv = nf(); w.bv = nf(); w.Wo = nf(); w.bo = nf();
    return w;
  };

  const float* vfeat   = nf();
  const float* efeat   = nf();
  const float* tempG   = nf();
  const float* structG = nf();
  const float* vtxW = nf(); const float* vtxb = nf();
  const float* nsaI = nf();
  MabW n0 = nmab(); MabW n1 = nmab();
  MabW ve = nmab();
  const float* elng = nf(); const float* elnb = nf();
  const float *sgW1 = nf(), *sgb1 = nf(), *sgg = nf(), *sgbn = nf(), *sgW2 = nf(), *sgb2 = nf();
  const float *tgW1 = nf(), *tgb1 = nf(), *tgg = nf(), *tgbn = nf(), *tgW2 = nf(), *tgb2 = nf();
  const float *mW1 = nf(), *mb1 = nf(), *mW2 = nf(), *mb2 = nf();
  const float* esaI = nf();
  MabW e0 = nmab(); MabW e1 = nmab();
  MabW ev = nmab();
  const float* nlng = nf(); const float* nlnb = nf();
  const int* edge_set = (const int*)d_in[p++];
  const int* node_set = (const int*)d_in[p++];

  // workspace carve
  float* ws = (float*)d_ws;
  size_t off = 0;
  auto alloc = [&](size_t n) { float* r = ws + off; off += n; return r; };
  float* featvL = alloc((size_t)DV * DD);
  float* feate  = alloc((size_t)DE * DD);
  float* q0n    = alloc(16 * DD);
  float* q0e    = alloc(16 * DD);
  float* t1     = alloc((size_t)DE * DD);
  float* t2     = alloc((size_t)DE * DD);
  float* sef    = alloc((size_t)DE * DD);
  float* tef    = alloc((size_t)DE * DD);
  float* catb   = alloc((size_t)DE * 2 * DD);
  float* hbuf   = alloc((size_t)DE * DH);
  float* fe2    = alloc((size_t)DE * DD);
  float* accum  = alloc(8);

  float* outF = (float*)d_out;
  const dim3 blk(256);

  // 1) feat_v = vfeat @ Wvtx + b
  k_gemm<<<dim3(DV / BM, 1), blk, 0, stream>>>(vfeat, DD, vtxW, DD, vtxb,
                                               featvL, DD, DV, DD, DD, 0);
  // 2) batch-invariant inducing-point queries
  k_gemm<<<dim3(1, 1), blk, 0, stream>>>(nsaI, DD, n0.Wq, DD, n0.bq, q0n, DD, 16, DD, DD, 0);
  k_gemm<<<dim3(1, 1), blk, 0, stream>>>(esaI, DD, e0.Wq, DD, e0.bq, q0e, DD, 16, DD, DD, 0);

  // 3) fused node->hyperedge encoder (ISAB + ve-MAB + layernorm), 1 wave/edge
  k_edge_encode<<<DE, 32, 0, stream>>>(featvL, efeat, edge_set, q0n,
                                       n0, n1, ve, elng, elnb, feate);

  // 4) dual GCNs on dense [E,E] adjacency: TDM-fed, double-buffered WMMA GEMMs
  // struct view
  k_gemm_tdm<<<dim3(DE / TBM), dim3(128), 0, stream>>>(structG, DE, feate, t1, DE, DE);
  k_gemm<<<dim3(DE / BM, 1), blk, 0, stream>>>(t1, DD, sgW1, DD, sgb1, t2, DD, DE, DD, DD, 0);
  k_layernorm<<<DE, 128, 0, stream>>>(t2, sgg, sgbn, t1);
  k_gemm_tdm<<<dim3(DE / TBM), dim3(128), 0, stream>>>(structG, DE, t1, t2, DE, DE);
  k_gemm<<<dim3(DE / BM, 1), blk, 0, stream>>>(t2, DD, sgW2, DD, sgb2, sef, DD, DE, DD, DD, 0);
  // temp view
  k_gemm_tdm<<<dim3(DE / TBM), dim3(128), 0, stream>>>(tempG, DE, feate, t1, DE, DE);
  k_gemm<<<dim3(DE / BM, 1), blk, 0, stream>>>(t1, DD, tgW1, DD, tgb1, t2, DD, DE, DD, DD, 0);
  k_layernorm<<<DE, 128, 0, stream>>>(t2, tgg, tgbn, t1);
  k_gemm_tdm<<<dim3(DE / TBM), dim3(128), 0, stream>>>(tempG, DE, t1, t2, DE, DE);
  k_gemm<<<dim3(DE / BM, 1), blk, 0, stream>>>(t2, DD, tgW2, DD, tgb2, tef, DD, DE, DD, DD, 0);

  // 5) SSL loss
  k_zero<<<1, 64, 0, stream>>>(accum, 8);
  k_cos<<<DE / 8, 256, 0, stream>>>(sef, tef, accum);
  k_loss<<<1, 32, 0, stream>>>(accum, outF);

  // 6) edge MLP on [struct || temp]
  k_concat<<<(DE * DD + 255) / 256, 256, 0, stream>>>(sef, tef, catb);
  k_gemm<<<dim3(DE / BM, DH / BN), blk, 0, stream>>>(catb, 2 * DD, mW1, DH, mb1,
                                                     hbuf, DH, DE, DH, 2 * DD, 1);
  k_gemm<<<dim3(DE / BM, 1), blk, 0, stream>>>(hbuf, DH, mW2, DD, mb2,
                                               fe2, DD, DE, DD, DH, 0);

  // 7) fused hyperedge->node decoder (ISAB + ev-MAB + layernorm), 1 wave/node
  k_node_decode<<<DV, 32, 0, stream>>>(featvL, fe2, node_set, q0e,
                                       e0, e1, ev, nlng, nlnb, outF);
}
